// HGTLayer_90683939488426
// MI455X (gfx1250) — matrix operand order repaired and checked by reading, compiled, not run on hardware
//
#include <hip/hip_runtime.h>
#include <math.h>

typedef __attribute__((ext_vector_type(16))) __bf16 bf16x16;
typedef __attribute__((ext_vector_type(8)))  float  v8f;

union BF16Frag { bf16x16 v; __bf16 e[16]; };

#define DIM 256
#define NHEAD 8
#define DK 32
#define INV_SQRT_DK 0.17677669529663687f

// ---------------------------------------------------------------------------
// CDNA5 async global->LDS copy (ASYNCcnt-tracked), saddr + 32-bit voffset form
// ---------------------------------------------------------------------------
#define ASYNC_LDS_B128(ldsaddr, voff, sbase)                                   \
    asm volatile("global_load_async_to_lds_b128 %0, %1, %2"                    \
                 :: "v"(ldsaddr), "v"(voff), "s"(sbase) : "memory")
#define WAIT_ASYNC_1() asm volatile("s_wait_asynccnt 0x1" ::: "memory")
#define WAIT_ASYNC_0() asm volatile("s_wait_asynccnt 0x0" ::: "memory")

__device__ __forceinline__ unsigned lds_addr_u32(const void* p) {
    return (unsigned)(uintptr_t)p;   // addrspace(3) offset lives in low 32 bits
}

// ---------------------------------------------------------------------------
// float atomic max via ordered-int trick (init must be -inf)
// ---------------------------------------------------------------------------
__device__ inline void atomicMaxFloat(float* addr, float val) {
    if (val >= 0.0f) atomicMax(reinterpret_cast<int*>(addr), __float_as_int(val));
    else             atomicMin(reinterpret_cast<unsigned int*>(addr), __float_as_uint(val));
}

// ---------------------------------------------------------------------------
// Fold per-head transform into projection weight:
//   Weff[d, h*32+j] = sum_i W[d, h*32+i] * trans[h, i, j]
// ---------------------------------------------------------------------------
__global__ __launch_bounds__(256) void make_eff_w(const float* __restrict__ W,
                                                  const float* __restrict__ trans,
                                                  float* __restrict__ Weff) {
    int d = blockIdx.x;           // 0..255
    int col = threadIdx.x;        // 0..255
    int h = col >> 5, j = col & 31;
    const float* wrow = W + d * DIM + h * DK;
    const float* tr = trans + h * DK * DK + j;
    float acc = 0.0f;
#pragma unroll 8
    for (int i = 0; i < DK; ++i) acc += wrow[i] * tr[i * DK];
    Weff[d * DIM + col] = acc;
}

__global__ __launch_bounds__(256) void make_eff_b(const float* __restrict__ b,
                                                  const float* __restrict__ trans,
                                                  float* __restrict__ beff) {
    int col = threadIdx.x;
    int h = col >> 5, j = col & 31;
    float acc = 0.0f;
#pragma unroll 8
    for (int i = 0; i < DK; ++i) acc += b[h * DK + i] * trans[h * DK * DK + i * DK + j];
    beff[col] = acc;
}

// ---------------------------------------------------------------------------
// Pre-swizzle a 256x256 fp32 weight into hi/lo bf16 B-matrix fragments.
// Fragment layout (16-bit B 32x16, contiguous k order, same as the A fragments):
//   linear idx = (((kchunk*16 + ntile)*32) + lane)*16 + pos
//   n = ntile*16 + (lane&15);  k = kchunk*32 + (lane>>4)*16 + pos
// ---------------------------------------------------------------------------
__global__ __launch_bounds__(256) void swizzle_w(const float* __restrict__ W,
                                                 __bf16* __restrict__ Whi,
                                                 __bf16* __restrict__ Wlo) {
    int idx = blockIdx.x * 256 + threadIdx.x;   // 0..65535
    int pos    = idx & 15;
    int lane   = (idx >> 4) & 31;
    int ntile  = (idx >> 9) & 15;
    int kchunk = idx >> 13;
    int k = kchunk * 32 + (lane >> 4) * 16 + pos;
    int n = ntile * 16 + (lane & 15);
    float v = W[k * DIM + n];
    __bf16 hi = (__bf16)v;
    __bf16 lo = (__bf16)(v - (float)hi);
    Whi[idx] = hi;
    Wlo[idx] = lo;
}

__global__ void init_neg_inf(float* __restrict__ p, int n) {
    int i = blockIdx.x * 256 + threadIdx.x;
    if (i < n) p[i] = __int_as_float((int)0xFF800000u);
}

// ---------------------------------------------------------------------------
// Split-precision bf16 WMMA GEMM:  Out[N x 256] = X[N x 256] @ W + bias
// Block: 256 threads = 8 waves. Tile: 32 rows x 256 cols.
// A tile is double-buffered in LDS via global_load_async_to_lds_b128.
// Wave w: m-tile = w&1, n-tiles = (w>>1)*4 .. +3.
// ---------------------------------------------------------------------------
#define AS_STRIDE 36   // floats; 144B rows keep 16B alignment for B128 async writes

__global__ __launch_bounds__(256) void gemm_bias(const float* __restrict__ X,
                                                 const bf16x16* __restrict__ Whi,
                                                 const bf16x16* __restrict__ Wlo,
                                                 const float* __restrict__ bias,
                                                 float* __restrict__ Out, int N) {
    __shared__ __align__(16) float As[2][32][AS_STRIDE];
    int tid = threadIdx.x;
    int wave = tid >> 5, lane = tid & 31;
    int rowBase = blockIdx.x * 32;
    int mt = wave & 1;
    int nt0 = (wave >> 1) * 4;
    int m = lane & 15, hlf = lane >> 4;

    // async staging mapping: thread -> (row r, 16B quad q) of the 32x32 fp32 tile
    int r = tid >> 3, q = tid & 7;
    int srow = rowBase + r;
    bool svalid = srow < N;
    unsigned ldsrow = lds_addr_u32(&As[0][0][0]) + (unsigned)(r * AS_STRIDE * 4 + q * 16);
    unsigned goff = (unsigned)srow * (DIM * 4) + (unsigned)(q * 16);
    unsigned long long xbase = (unsigned long long)X;

    if (rowBase + 32 > N) {  // tail block: pre-zero both buffers
        for (int i = tid; i < 2 * 32 * AS_STRIDE; i += 256) (&As[0][0][0])[i] = 0.0f;
        __syncthreads();
    }
    if (svalid) ASYNC_LDS_B128(ldsrow, goff, xbase);   // prologue: chunk 0

    v8f zero = {0.f, 0.f, 0.f, 0.f, 0.f, 0.f, 0.f, 0.f};
    v8f acc[4] = {zero, zero, zero, zero};

    for (int c = 0; c < 8; ++c) {
        if (c < 7) {
            if (svalid)
                ASYNC_LDS_B128(ldsrow + (unsigned)(((c + 1) & 1) * 32 * AS_STRIDE * 4),
                               goff + (unsigned)((c + 1) * 128), xbase);
            WAIT_ASYNC_1();
        } else {
            WAIT_ASYNC_0();
        }
        __syncthreads();   // all waves' async writes landed

        int buf = c & 1;
        BF16Frag ahi, alo;
#pragma unroll
        for (int p = 0; p < 16; ++p) {
            int kl = hlf * 16 + p;  // contiguous k order, same as swizzle_w
            float x = As[buf][mt * 16 + m][kl];
            __bf16 hi = (__bf16)x;
            ahi.e[p] = hi;
            alo.e[p] = (__bf16)(x - (float)hi);
        }
#pragma unroll
        for (int t = 0; t < 4; ++t) {
            int fi = (c * 16 + nt0 + t) * 32 + lane;
            bf16x16 bh = Whi[fi];
            bf16x16 bl = Wlo[fi];
            acc[t] = __builtin_amdgcn_wmma_f32_16x16x32_bf16(false, ahi.v, false, bh, (short)0, acc[t], false, false);
            acc[t] = __builtin_amdgcn_wmma_f32_16x16x32_bf16(false, ahi.v, false, bl, (short)0, acc[t], false, false);
            acc[t] = __builtin_amdgcn_wmma_f32_16x16x32_bf16(false, alo.v, false, bh, (short)0, acc[t], false, false);
        }
        __syncthreads();   // done reading this buffer before it is re-filled
    }

#pragma unroll
    for (int t = 0; t < 4; ++t) {
        int col = (nt0 + t) * 16 + m;
        float bv = bias[col];
#pragma unroll
        for (int rr = 0; rr < 8; ++rr) {
            int row = rowBase + mt * 16 + hlf * 8 + rr;
            if (row < N) Out[(size_t)row * DIM + col] = acc[t][rr] + bv;
        }
    }
}

// ---------------------------------------------------------------------------
// Edge scores: one wave per edge. score[e,h] = (q[dst,h,:].k[src,h,:])*pri[h]/sqrt(dk)
// Also atomic-max into per-dst per-head smax.
// ---------------------------------------------------------------------------
__global__ __launch_bounds__(256) void edge_score(const float* __restrict__ Q,
                                                  const float* __restrict__ K,
                                                  const int* __restrict__ src,
                                                  const int* __restrict__ dst,
                                                  const float* __restrict__ pri,
                                                  float* __restrict__ score,
                                                  float* __restrict__ smax, int E) {
    int e = blockIdx.x * 8 + (threadIdx.x >> 5);
    if (e >= E) return;
    int lane = threadIdx.x & 31;
    int sn = src[e], dn = dst[e];
    const float4* q4 = reinterpret_cast<const float4*>(Q + (size_t)dn * DIM);
    const float4* k4 = reinterpret_cast<const float4*>(K + (size_t)sn * DIM);
    float4 qa = q4[lane * 2], qb = q4[lane * 2 + 1];
    float4 ka = k4[lane * 2], kb = k4[lane * 2 + 1];
    float p = qa.x * ka.x + qa.y * ka.y + qa.z * ka.z + qa.w * ka.w +
              qb.x * kb.x + qb.y * kb.y + qb.z * kb.z + qb.w * kb.w;
    p += __shfl_xor(p, 1);
    p += __shfl_xor(p, 2);
    if ((lane & 3) == 0) {
        int h = lane >> 2;
        float sc = p * pri[h] * INV_SQRT_DK;
        score[(size_t)e * NHEAD + h] = sc;
        atomicMaxFloat(smax + (size_t)dn * NHEAD + h, sc);
    }
}

// ---------------------------------------------------------------------------
// ex = exp(score - smax[dst]); ssum[dst,h] += ex; t[dst,:] += v[src,:]*ex_head
// (normalization by ssum deferred to dst_update)
// ---------------------------------------------------------------------------
__global__ __launch_bounds__(256) void edge_aggregate(const float* __restrict__ V,
                                                      const float* __restrict__ score,
                                                      const float* __restrict__ smax,
                                                      const int* __restrict__ src,
                                                      const int* __restrict__ dst,
                                                      float* __restrict__ ssum,
                                                      float* __restrict__ t, int E) {
    int e = blockIdx.x * 8 + (threadIdx.x >> 5);
    if (e >= E) return;
    int lane = threadIdx.x & 31;
    int sn = src[e], dn = dst[e];
    float ex = 0.0f;
    if (lane < NHEAD) {
        ex = __expf(score[(size_t)e * NHEAD + lane] - smax[(size_t)dn * NHEAD + lane]);
        atomicAdd(ssum + (size_t)dn * NHEAD + lane, ex);
    }
    float exh = __shfl(ex, lane >> 2);   // head of elements [lane*8, lane*8+8)
    const float4* v4 = reinterpret_cast<const float4*>(V + (size_t)sn * DIM);
    float4 va = v4[lane * 2], vb = v4[lane * 2 + 1];
    float* td = t + (size_t)dn * DIM + lane * 8;
    atomicAdd(td + 0, va.x * exh);
    atomicAdd(td + 1, va.y * exh);
    atomicAdd(td + 2, va.z * exh);
    atomicAdd(td + 3, va.w * exh);
    atomicAdd(td + 4, vb.x * exh);
    atomicAdd(td + 5, vb.y * exh);
    atomicAdd(td + 6, vb.z * exh);
    atomicAdd(td + 7, vb.w * exh);
}

// ---------------------------------------------------------------------------
// dst update: out = LayerNorm( (t/ssum @ Wa + ba)*alpha + h*(1-alpha) )
// Async double-buffered A staging (raw t); per-row 1/ssum applied at
// fragment-build time. LN fused via LDS row reductions.
// ---------------------------------------------------------------------------
__global__ __launch_bounds__(256) void dst_update(const float* __restrict__ T,
                                                  const float* __restrict__ ssum,
                                                  const float* __restrict__ Hin,
                                                  const bf16x16* __restrict__ Whi,
                                                  const bf16x16* __restrict__ Wlo,
                                                  const float* __restrict__ ba,
                                                  const float* __restrict__ skip,
                                                  const float* __restrict__ g,
                                                  const float* __restrict__ bln,
                                                  float* __restrict__ Out, int N) {
    __shared__ __align__(16) float As[2][32][AS_STRIDE];
    __shared__ float XT[32][257];
    __shared__ float2 red[32][8];
    __shared__ float2 stats[32];

    int tid = threadIdx.x;
    int wave = tid >> 5, lane = tid & 31;
    int rowBase = blockIdx.x * 32;
    int mt = wave & 1;
    int nt0 = (wave >> 1) * 4;
    int m = lane & 15, hlf = lane >> 4;
    float alpha = 1.0f / (1.0f + __expf(-skip[0]));

    int r = tid >> 3, q = tid & 7;
    int srow = rowBase + r;
    bool svalid = srow < N;
    unsigned ldsrow = lds_addr_u32(&As[0][0][0]) + (unsigned)(r * AS_STRIDE * 4 + q * 16);
    unsigned goff = (unsigned)srow * (DIM * 4) + (unsigned)(q * 16);
    unsigned long long tbase = (unsigned long long)T;

    int myrow = rowBase + mt * 16 + m;          // row this lane's A fragment belongs to
    const float* ssum_row = ssum + (size_t)(myrow < N ? myrow : 0) * NHEAD;

    if (rowBase + 32 > N) {
        for (int i = tid; i < 2 * 32 * AS_STRIDE; i += 256) (&As[0][0][0])[i] = 0.0f;
        __syncthreads();
    }
    if (svalid) ASYNC_LDS_B128(ldsrow, goff, tbase);

    v8f zero = {0.f, 0.f, 0.f, 0.f, 0.f, 0.f, 0.f, 0.f};
    v8f acc[4] = {zero, zero, zero, zero};

    for (int c = 0; c < 8; ++c) {
        if (c < 7) {
            if (svalid)
                ASYNC_LDS_B128(ldsrow + (unsigned)(((c + 1) & 1) * 32 * AS_STRIDE * 4),
                               goff + (unsigned)((c + 1) * 128), tbase);
            WAIT_ASYNC_1();
        } else {
            WAIT_ASYNC_0();
        }
        __syncthreads();

        // per-row softmax normalization for this k-chunk's head (head == c)
        float sv = (myrow < N) ? ssum_row[c] : 0.0f;
        float scale = (sv > 0.0f) ? (1.0f / sv) : 0.0f;

        int buf = c & 1;
        BF16Frag ahi, alo;
#pragma unroll
        for (int p = 0; p < 16; ++p) {
            int kl = hlf * 16 + p;
            float x = As[buf][mt * 16 + m][kl] * scale;
            __bf16 hi = (__bf16)x;
            ahi.e[p] = hi;
            alo.e[p] = (__bf16)(x - (float)hi);
        }
#pragma unroll
        for (int t = 0; t < 4; ++t) {
            int fi = (c * 16 + nt0 + t) * 32 + lane;
            bf16x16 bh = Whi[fi];
            bf16x16 bl = Wlo[fi];
            acc[t] = __builtin_amdgcn_wmma_f32_16x16x32_bf16(false, ahi.v, false, bh, (short)0, acc[t], false, false);
            acc[t] = __builtin_amdgcn_wmma_f32_16x16x32_bf16(false, ahi.v, false, bl, (short)0, acc[t], false, false);
            acc[t] = __builtin_amdgcn_wmma_f32_16x16x32_bf16(false, alo.v, false, bh, (short)0, acc[t], false, false);
        }
        __syncthreads();
    }

    // skip-gated pre-LN values into LDS
#pragma unroll
    for (int t = 0; t < 4; ++t) {
        int col = (nt0 + t) * 16 + m;
        float bv = ba[col];
#pragma unroll
        for (int rr = 0; rr < 8; ++rr) {
            int rloc = mt * 16 + hlf * 8 + rr;
            int row = rowBase + rloc;
            float hv = (row < N) ? Hin[(size_t)row * DIM + col] : 0.0f;
            XT[rloc][col] = (acc[t][rr] + bv) * alpha + hv * (1.0f - alpha);
        }
    }
    __syncthreads();

    // LayerNorm: 8 threads per row, 32 cols each
    int rloc = tid >> 3, seg = tid & 7;
    float sum = 0.0f, sq = 0.0f;
    for (int c = seg * 32; c < seg * 32 + 32; ++c) {
        float x = XT[rloc][c];
        sum += x; sq += x * x;
    }
    red[rloc][seg] = make_float2(sum, sq);
    __syncthreads();
    if (tid < 32) {
        float S = 0.0f, Q = 0.0f;
#pragma unroll
        for (int j = 0; j < 8; ++j) { S += red[tid][j].x; Q += red[tid][j].y; }
        float mean = S * (1.0f / DIM);
        float var = Q * (1.0f / DIM) - mean * mean;
        stats[tid] = make_float2(mean, rsqrtf(var + 1e-5f));
    }
    __syncthreads();
    int row = rowBase + rloc;
    if (row < N) {
        float mean = stats[rloc].x, rstd = stats[rloc].y;
        for (int c = seg * 32; c < seg * 32 + 32; ++c) {
            float x = XT[rloc][c];
            Out[(size_t)row * DIM + c] = (x - mean) * rstd * g[c] + bln[c];
        }
    }
}

// ---------------------------------------------------------------------------
// Host-side orchestration
// ---------------------------------------------------------------------------
extern "C" void kernel_launch(void* const* d_in, const int* in_sizes, int n_in,
                              void* d_out, int out_size, void* d_ws, size_t ws_size,
                              hipStream_t stream) {
    const float* h_user = (const float*)d_in[0];
    const float* h_item = (const float*)d_in[1];
    const int* src_ui = (const int*)d_in[2];
    const int* dst_ui = (const int*)d_in[3];
    const int* src_iu = (const int*)d_in[4];
    const int* dst_iu = (const int*)d_in[5];
    const float* Wk_user = (const float*)d_in[6];  const float* bk_user = (const float*)d_in[7];
    const float* Wq_user = (const float*)d_in[8];  const float* bq_user = (const float*)d_in[9];
    const float* Wv_user = (const float*)d_in[10]; const float* bv_user = (const float*)d_in[11];
    const float* Wa_user = (const float*)d_in[12]; const float* ba_user = (const float*)d_in[13];
    const float* g_user  = (const float*)d_in[14]; const float* bln_user = (const float*)d_in[15];
    const float* skip_user = (const float*)d_in[16];
    const float* Wk_item = (const float*)d_in[17]; const float* bk_item = (const float*)d_in[18];
    const float* Wq_item = (const float*)d_in[19]; const float* bq_item = (const float*)d_in[20];
    const float* Wv_item = (const float*)d_in[21]; const float* bv_item = (const float*)d_in[22];
    const float* Wa_item = (const float*)d_in[23]; const float* ba_item = (const float*)d_in[24];
    const float* g_item  = (const float*)d_in[25]; const float* bln_item = (const float*)d_in[26];
    const float* skip_item = (const float*)d_in[27];
    const float* pri_ui = (const float*)d_in[28];
    const float* att_ui = (const float*)d_in[29];
    const float* msg_ui = (const float*)d_in[30];
    const float* pri_iu = (const float*)d_in[31];
    const float* att_iu = (const float*)d_in[32];
    const float* msg_iu = (const float*)d_in[33];

    const int Nu = in_sizes[0] / DIM;
    const int Ni = in_sizes[1] / DIM;
    const int E  = in_sizes[2];

    char* ws = (char*)d_ws;
    size_t off = 0;
    auto alloc = [&](size_t bytes) -> char* {
        char* p = ws + off;
        off += (bytes + 255) & ~(size_t)255;
        return p;
    };

    const size_t WB = (size_t)DIM * DIM * sizeof(float);    // 256 KB fp32
    const size_t FB = (size_t)DIM * DIM * sizeof(__bf16);   // 128 KB per half

    float* Weff_k_user = (float*)alloc(WB);
    float* Weff_v_user = (float*)alloc(WB);
    float* Weff_k_item = (float*)alloc(WB);
    float* Weff_v_item = (float*)alloc(WB);

    __bf16* f_hi[8];  __bf16* f_lo[8];   // ku, vu, qu, au, ki, vi, qi, ai
    for (int i = 0; i < 8; ++i) { f_hi[i] = (__bf16*)alloc(FB); f_lo[i] = (__bf16*)alloc(FB); }

    float* beff_k_user = (float*)alloc(DIM * sizeof(float));
    float* beff_v_user = (float*)alloc(DIM * sizeof(float));
    float* beff_k_item = (float*)alloc(DIM * sizeof(float));
    float* beff_v_item = (float*)alloc(DIM * sizeof(float));

    float* Ku = (float*)alloc((size_t)Nu * DIM * sizeof(float));
    float* Vu = (float*)alloc((size_t)Nu * DIM * sizeof(float));
    float* Qu = (float*)alloc((size_t)Nu * DIM * sizeof(float));
    float* Ki = (float*)alloc((size_t)Ni * DIM * sizeof(float));
    float* Vi = (float*)alloc((size_t)Ni * DIM * sizeof(float));
    float* Qi = (float*)alloc((size_t)Ni * DIM * sizeof(float));

    float* score_ui = (float*)alloc((size_t)E * NHEAD * sizeof(float));
    float* score_iu = (float*)alloc((size_t)E * NHEAD * sizeof(float));

    float* smax = (float*)alloc((size_t)(Ni + Nu) * NHEAD * sizeof(float));
    float* smax_ui = smax;                        // dst = item
    float* smax_iu = smax + (size_t)Ni * NHEAD;   // dst = user

    size_t zfloats = (size_t)(Ni + Nu) * NHEAD + (size_t)(Ni + Nu) * DIM;
    float* zbase = (float*)alloc(zfloats * sizeof(float));
    float* ssum_ui = zbase;
    float* ssum_iu = zbase + (size_t)Ni * NHEAD;
    float* t_item  = zbase + (size_t)(Ni + Nu) * NHEAD;
    float* t_user  = t_item + (size_t)Ni * DIM;

    // --- init accumulators ---
    hipMemsetAsync(zbase, 0, zfloats * sizeof(float), stream);
    {
        int n = (Ni + Nu) * NHEAD;
        init_neg_inf<<<(n + 255) / 256, 256, 0, stream>>>(smax, n);
    }

    // --- fold head transforms into weights, build effective biases ---
    make_eff_w<<<256, 256, 0, stream>>>(Wk_user, att_ui, Weff_k_user);
    make_eff_w<<<256, 256, 0, stream>>>(Wv_user, msg_ui, Weff_v_user);
    make_eff_w<<<256, 256, 0, stream>>>(Wk_item, att_iu, Weff_k_item);
    make_eff_w<<<256, 256, 0, stream>>>(Wv_item, msg_iu, Weff_v_item);
    make_eff_b<<<1, 256, 0, stream>>>(bk_user, att_ui, beff_k_user);
    make_eff_b<<<1, 256, 0, stream>>>(bv_user, msg_ui, beff_v_user);
    make_eff_b<<<1, 256, 0, stream>>>(bk_item, att_iu, beff_k_item);
    make_eff_b<<<1, 256, 0, stream>>>(bv_item, msg_iu, beff_v_item);

    // --- split + swizzle weights into WMMA B-fragment layout ---
    swizzle_w<<<256, 256, 0, stream>>>(Weff_k_user, f_hi[0], f_lo[0]);
    swizzle_w<<<256, 256, 0, stream>>>(Weff_v_user, f_hi[1], f_lo[1]);
    swizzle_w<<<256, 256, 0, stream>>>(Wq_user,     f_hi[2], f_lo[2]);
    swizzle_w<<<256, 256, 0, stream>>>(Wa_user,     f_hi[3], f_lo[3]);
    swizzle_w<<<256, 256, 0, stream>>>(Weff_k_item, f_hi[4], f_lo[4]);
    swizzle_w<<<256, 256, 0, stream>>>(Weff_v_item, f_hi[5], f_lo[5]);
    swizzle_w<<<256, 256, 0, stream>>>(Wq_item,     f_hi[6], f_lo[6]);
    swizzle_w<<<256, 256, 0, stream>>>(Wa_item,     f_hi[7], f_lo[7]);

    // --- K/V/Q projections (split-precision bf16 WMMA, async LDS staging) ---
    int gU = (Nu + 31) / 32, gI = (Ni + 31) / 32;
    gemm_bias<<<gU, 256, 0, stream>>>(h_user, (const bf16x16*)f_hi[0], (const bf16x16*)f_lo[0], beff_k_user, Ku, Nu);
    gemm_bias<<<gU, 256, 0, stream>>>(h_user, (const bf16x16*)f_hi[1], (const bf16x16*)f_lo[1], beff_v_user, Vu, Nu);
    gemm_bias<<<gU, 256, 0, stream>>>(h_user, (const bf16x16*)f_hi[2], (const bf16x16*)f_lo[2], bq_user,     Qu, Nu);
    gemm_bias<<<gI, 256, 0, stream>>>(h_item, (const bf16x16*)f_hi[4], (const bf16x16*)f_lo[4], beff_k_item, Ki, Ni);
    gemm_bias<<<gI, 256, 0, stream>>>(h_item, (const bf16x16*)f_hi[5], (const bf16x16*)f_lo[5], beff_v_item, Vi, Ni);
    gemm_bias<<<gI, 256, 0, stream>>>(h_item, (const bf16x16*)f_hi[6], (const bf16x16*)f_lo[6], bq_item,     Qi, Ni);

    // --- edge attention ---
    int gE = (E + 7) / 8;
    edge_score<<<gE, 256, 0, stream>>>(Qi, Ku, src_ui, dst_ui, pri_ui, score_ui, smax_ui, E);
    edge_score<<<gE, 256, 0, stream>>>(Qu, Ki, src_iu, dst_iu, pri_iu, score_iu, smax_iu, E);
    edge_aggregate<<<gE, 256, 0, stream>>>(Vu, score_ui, smax_ui, src_ui, dst_ui, ssum_ui, t_item, E);
    edge_aggregate<<<gE, 256, 0, stream>>>(Vi, score_iu, smax_iu, src_iu, dst_iu, ssum_iu, t_user, E);

    // --- dst update + LayerNorm (fused WMMA GEMM) ---
    float* out_user = (float*)d_out;
    float* out_item = (float*)d_out + (size_t)Nu * DIM;
    dst_update<<<gU, 256, 0, stream>>>(t_user, ssum_iu, h_user,
                                       (const bf16x16*)f_hi[3], (const bf16x16*)f_lo[3],
                                       ba_user, skip_user, g_user, bln_user, out_user, Nu);
    dst_update<<<gI, 256, 0, stream>>>(t_item, ssum_ui, h_item,
                                       (const bf16x16*)f_hi[7], (const bf16x16*)f_lo[7],
                                       ba_item, skip_item, g_item, bln_item, out_item, Ni);
}
